// X_vector_85538568667488
// MI455X (gfx1250) — matrix operand order repaired and checked by reading, not compiled, not run on hardware
//
#include <hip/hip_runtime.h>
#include <hip/hip_bf16.h>

// ---------------------------------------------------------------------------
// X-vector TDNN forward for MI455X (gfx1250, wave32, WMMA + async LDS DMA).
// Conv/FC layers are implicit GEMMs on v_wmma_f32_16x16x32_bf16.
// 4-wave workgroups share B tiles staged via global_load_async_to_lds_b128
// (ASYNCcnt-tracked, double-buffered) to cut L2 weight traffic 4x.
// ---------------------------------------------------------------------------

typedef __attribute__((ext_vector_type(16))) __bf16 v16bf;
typedef __attribute__((ext_vector_type(8)))  float  v8f;

union FragU { int4 q[2]; v16bf v; };

__device__ __forceinline__ __bf16 f2bf(float f) {
  union { float f; unsigned u; } in; in.f = f;
  unsigned u = in.u + 0x7FFFu + ((in.u >> 16) & 1u);   // round-to-nearest-even
  union { unsigned short s; __bf16 b; } out; out.s = (unsigned short)(u >> 16);
  return out.b;
}
__device__ __forceinline__ float bf2f(__bf16 b) {
  union { unsigned short s; __bf16 b; } in; in.b = b;
  union { unsigned u; float f; } out; out.u = ((unsigned)in.s) << 16;
  return out.f;
}

// LDS B tile: 64 cols x 64 K, row stride 72 bf16 (144B) to spread banks.
#define BT_STRIDE 72
#define BT_ELEMS  (64 * BT_STRIDE)

// Stage one 64x64-K B tile into LDS: 128 threads x 4 chunks of 16B.
// Chunk c = tid + 128*i -> col = c>>3, 16B-piece = c&7. Async, ASYNCcnt += 4.
__device__ __forceinline__ void async_stage_B(const __bf16* __restrict__ Wnb,
                                              int Kpad, __bf16* bufBase,
                                              int tid, int k0)
{
#pragma unroll
  for (int i = 0; i < 4; ++i) {
    const int c   = tid + 128 * i;
    const int col = c >> 3;
    const int o8  = c & 7;
    const __bf16* g = Wnb + (size_t)col * Kpad + k0 + o8 * 8;
    const unsigned lds = (unsigned)(size_t)(bufBase + col * BT_STRIDE + o8 * 8);
    asm volatile("global_load_async_to_lds_b128 %0, %1, off"
                 :: "v"(lds), "v"(g) : "memory");
  }
}

// ---------------------------------------------------------------------------
// Implicit-GEMM conv / linear layer.
//   A : bf16 activations [B, T_in, C_in]; GEMM row m=(b,t) starts at
//       (b*T_in + t + start)*C_in, spans Kpad contiguous bf16 (Kpad % 64 == 0).
//   W : bf16 weights [N_pad][K_pad] row-major, zero padded.
// Block: 128 threads = 4 waves; tile 128(M) x 64(N); wave tile 32x64.
// ---------------------------------------------------------------------------
__global__ __launch_bounds__(128)
void gemm_wmma(const __bf16* __restrict__ A, const __bf16* __restrict__ W,
               const float* __restrict__ bias,
               __bf16* __restrict__ outBf, float* __restrict__ outF32,
               int Mtot, int Nreal, int Kpad, int Tin, int Tout, int Cin,
               int start, int ldo, int relu)
{
  __shared__ __bf16 Bt[2][BT_ELEMS];

  const int tid  = threadIdx.x;
  const int wave = tid >> 5;
  const int lane = tid & 31;
  const int half = lane >> 4;
  const int lr   = lane & 15;
  const int nb   = blockIdx.y;

  const __bf16* Wnb = W + (size_t)(nb * 64) * Kpad;

  // Two A fragment rows per wave (clamped so tail blocks stay in-bounds).
  int m0 = blockIdx.x * 128 + wave * 32 + lr;
  int m1 = m0 + 16;
  if (m0 > Mtot - 1) m0 = Mtot - 1;
  if (m1 > Mtot - 1) m1 = Mtot - 1;
  const int b0 = m0 / Tout, t0 = m0 - b0 * Tout;
  const int b1 = m1 / Tout, t1 = m1 - b1 * Tout;
  const __bf16* ap0 = A + (size_t)(b0 * Tin + t0 + start) * Cin + 8 * half;
  const __bf16* ap1 = A + (size_t)(b1 * Tin + t1 + start) * Cin + 8 * half;

  v8f acc[2][4] = {};

  const int nsteps = Kpad >> 6;          // 64-K steps
  async_stage_B(Wnb, Kpad, &Bt[0][0], tid, 0);

  for (int it = 0; it < nsteps; ++it) {
    const int k64 = it << 6;
    __bf16* cur = &Bt[it & 1][0];
    __bf16* nxt = &Bt[(it & 1) ^ 1][0];

    __syncthreads();                     // everyone done reading old 'nxt' data
    if (it + 1 < nsteps) {
      async_stage_B(Wnb, Kpad, nxt, tid, k64 + 64);
      asm volatile("s_wait_asynccnt 0x4" ::: "memory");  // step 'it' resident
    } else {
      asm volatile("s_wait_asynccnt 0x0" ::: "memory");
    }
    __syncthreads();                     // all waves' 'cur' portions visible

    // Load all A fragments for both 32-K substeps up front (overlap ds reads).
    FragU a[2][2];
#pragma unroll
    for (int s = 0; s < 2; ++s) {
      const int kk = k64 + s * 32;
      a[s][0].q[0] = *(const int4*)(ap0 + kk);
      a[s][0].q[1] = *(const int4*)(ap0 + kk + 16);
      a[s][1].q[0] = *(const int4*)(ap1 + kk);
      a[s][1].q[1] = *(const int4*)(ap1 + kk + 16);
    }
#pragma unroll
    for (int s = 0; s < 2; ++s) {
      const __bf16* bbase = cur + s * 32 + 8 * half;
      FragU bf[4];
#pragma unroll
      for (int j = 0; j < 4; ++j) {
        const __bf16* p = bbase + (j * 16 + lr) * BT_STRIDE;
        bf[j].q[0] = *(const int4*)(p);
        bf[j].q[1] = *(const int4*)(p + 16);
      }
      // Pass 1 (first use of each bf[j]) ... all four must stay live ...
#pragma unroll
      for (int j = 0; j < 4; ++j)
        acc[0][j] = __builtin_amdgcn_wmma_f32_16x16x32_bf16(
            false, a[s][0].v, false, bf[j].v, (short)0, acc[0][j], false, false);
      // ... Pass 2 (second use): forces 4 distinct register sets, runs
      // wait-free, and lets next substep's ds loads hide under the WMMAs.
#pragma unroll
      for (int j = 0; j < 4; ++j)
        acc[1][j] = __builtin_amdgcn_wmma_f32_16x16x32_bf16(
            false, a[s][1].v, false, bf[j].v, (short)0, acc[1][j], false, false);
    }
  }

  // D layout: VGPR r -> row (+ s*16 + 8*half + r), col (nb*64 + j*16 + lr)
#pragma unroll
  for (int s = 0; s < 2; ++s) {
#pragma unroll
    for (int j = 0; j < 4; ++j) {
      const int col = nb * 64 + j * 16 + lr;
      if (col >= Nreal) continue;
      const float bv = bias ? bias[col] : 0.0f;
#pragma unroll
      for (int r = 0; r < 8; ++r) {
        const int row = blockIdx.x * 128 + wave * 32 + s * 16 + 8 * half + r;
        if (row >= Mtot) continue;
        float v = acc[s][j][r] + bv;
        if (relu && v < 0.0f) v = 0.0f;
        const size_t o = (size_t)row * ldo + col;
        if (outBf)  outBf[o]  = f2bf(v);
        if (outF32) outF32[o] = v;
      }
    }
  }
}

// ---- prep: x [B, 40, T] fp32 -> h0 [B, T, 40] bf16, zero tail pad ----------
__global__ void prep_x_kernel(const float* __restrict__ x, __bf16* __restrict__ h0,
                              int total_real, int total_alloc)
{
  int idx = blockIdx.x * blockDim.x + threadIdx.x;
  if (idx >= total_alloc) return;
  if (idx >= total_real) { h0[idx] = f2bf(0.0f); return; }
  int c = idx % 40;
  int t = (idx / 40) % 1000;
  int b = idx / (40 * 1000);
  h0[idx] = f2bf(x[((size_t)(b * 40 + c)) * 1000 + t]);
}

// ---- prep: conv kernel [Cout, Cin, W] fp32 -> Wt [Npad][Kpad] bf16 ---------
__global__ void prep_conv_kernel(const float* __restrict__ k, __bf16* __restrict__ wt,
                                 int Cout, int Cin, int W, int Npad, int Kpad)
{
  int idx = blockIdx.x * blockDim.x + threadIdx.x;
  if (idx >= Npad * Kpad) return;
  int kk = idx % Kpad;
  int n  = idx / Kpad;
  float v = 0.0f;
  if (n < Cout && kk < W * Cin) {
    int w  = kk / Cin;
    int ci = kk - w * Cin;
    v = k[((size_t)n * Cin + ci) * W + w];
  }
  wt[idx] = f2bf(v);
}

// ---- prep: torch Linear weight [Nreal, Kreal] fp32 -> [Npad][Kpad] bf16 ----
__global__ void prep_lin_kernel(const float* __restrict__ w, __bf16* __restrict__ wt,
                                int Nreal, int Kreal, int Npad, int Kpad)
{
  int idx = blockIdx.x * blockDim.x + threadIdx.x;
  if (idx >= Npad * Kpad) return;
  int k = idx % Kpad;
  int n = idx / Kpad;
  float v = (n < Nreal && k < Kreal) ? w[(size_t)n * Kreal + k] : 0.0f;
  wt[idx] = f2bf(v);
}

// ---- stats pooling: mean + unbiased std over 984 frames --------------------
__global__ void stats_kernel(const __bf16* __restrict__ h5, __bf16* __restrict__ st)
{
  const int T5 = 984, C5 = 1500, LDS_ = 3008;
  int idx = blockIdx.x * blockDim.x + threadIdx.x;
  if (idx >= 32 * 1504) return;
  int b = idx / 1504;
  int j = idx - b * 1504;
  if (j >= 1500) {                       // zero the 8 K-pad columns
    int p = j - 1500;
    st[(size_t)b * LDS_ + 3000 + p] = f2bf(0.0f);
    st[(size_t)b * LDS_ + 3004 + p] = f2bf(0.0f);
    return;
  }
  const __bf16* p = h5 + (size_t)b * T5 * C5 + j;
  float s = 0.0f;
  for (int t = 0; t < T5; ++t) s += bf2f(p[(size_t)t * C5]);
  float mean = s / (float)T5;
  float ss = 0.0f;
  for (int t = 0; t < T5; ++t) { float d = bf2f(p[(size_t)t * C5]) - mean; ss += d * d; }
  float sd = sqrtf(ss / (float)(T5 - 1));
  st[(size_t)b * LDS_ + j]        = f2bf(mean);
  st[(size_t)b * LDS_ + 1500 + j] = f2bf(sd);
}

// ---------------------------------------------------------------------------
extern "C" void kernel_launch(void* const* d_in, const int* in_sizes, int n_in,
                              void* d_out, int out_size, void* d_ws, size_t ws_size,
                              hipStream_t stream)
{
  (void)in_sizes; (void)n_in; (void)out_size; (void)ws_size;
  const float* x  = (const float*)d_in[0];
  const float* k1 = (const float*)d_in[1];  const float* c1 = (const float*)d_in[2];
  const float* k2 = (const float*)d_in[3];  const float* c2 = (const float*)d_in[4];
  const float* k3 = (const float*)d_in[5];  const float* c3 = (const float*)d_in[6];
  const float* k4 = (const float*)d_in[7];  const float* c4 = (const float*)d_in[8];
  const float* k5 = (const float*)d_in[9];  const float* c5 = (const float*)d_in[10];
  const float* w1 = (const float*)d_in[11]; const float* d1 = (const float*)d_in[12];
  const float* w2 = (const float*)d_in[13]; const float* d2 = (const float*)d_in[14];
  const float* wl = (const float*)d_in[15]; const float* dl = (const float*)d_in[16];
  float* emb_out  = (float*)d_out;                 // [32,512]
  float* prob_out = (float*)d_out + 32 * 512;      // [32,1000]

  // --- carve workspace (bf16 buffers, 256B aligned) ---
  char* base = (char*)d_ws;
  size_t off = 0;
  auto carve = [&](size_t elems) -> __bf16* {
    __bf16* p = (__bf16*)(base + off);
    off += ((elems * sizeof(__bf16)) + 255) & ~(size_t)255;
    return p;
  };
  __bf16* H0  = carve(1280000 + 512);              // [32,1000,40] + pad
  __bf16* H1  = carve((size_t)32 * 996 * 512);
  __bf16* H2  = carve((size_t)32 * 992 * 512);
  __bf16* H3  = carve((size_t)32 * 986 * 512);
  __bf16* H4  = carve((size_t)32 * 985 * 512);
  __bf16* H5  = carve((size_t)32 * 984 * 1500);
  __bf16* ST  = carve((size_t)32 * 3008);
  __bf16* E1  = carve((size_t)32 * 512);
  __bf16* EM  = carve((size_t)32 * 512);
  __bf16* W1T = carve((size_t)512  * 256);         // K=5*40=200 -> 256
  __bf16* W2T = carve((size_t)512  * 2560);
  __bf16* W3T = carve((size_t)512  * 3584);
  __bf16* W4T = carve((size_t)512  * 512);
  __bf16* W5T = carve((size_t)1536 * 512);         // N=1500 -> 1536
  __bf16* F1T = carve((size_t)512  * 3008);        // K=3000 -> 3008
  __bf16* F2T = carve((size_t)512  * 512);
  __bf16* FLT = carve((size_t)1024 * 512);         // N=1000 -> 1024

  const int PB = 256;
  auto g1 = [&](int n) { return (n + PB - 1) / PB; };

  // --- weight / input prep ---
  prep_x_kernel   <<<g1(1280512), PB, 0, stream>>>(x, H0, 1280000, 1280512);
  prep_conv_kernel<<<g1(512 * 256),  PB, 0, stream>>>(k1, W1T, 512,  40, 5, 512,  256);
  prep_conv_kernel<<<g1(512 * 2560), PB, 0, stream>>>(k2, W2T, 512, 512, 5, 512, 2560);
  prep_conv_kernel<<<g1(512 * 3584), PB, 0, stream>>>(k3, W3T, 512, 512, 7, 512, 3584);
  prep_conv_kernel<<<g1(512 * 512),  PB, 0, stream>>>(k4, W4T, 512, 512, 1, 512,  512);
  prep_conv_kernel<<<g1(1536 * 512), PB, 0, stream>>>(k5, W5T, 1500, 512, 1, 1536, 512);
  prep_lin_kernel <<<g1(512 * 3008), PB, 0, stream>>>(w1, F1T, 512, 3000, 512, 3008);
  prep_lin_kernel <<<g1(512 * 512),  PB, 0, stream>>>(w2, F2T, 512,  512, 512,  512);
  prep_lin_kernel <<<g1(1024 * 512), PB, 0, stream>>>(wl, FLT, 1000, 512, 1024, 512);

  // --- TDNN layers (grid.x = ceil(M/128), 128 threads = 4 waves) ---
  // args: A, W, bias, outBf, outF32, Mtot, Nreal, Kpad, Tin, Tout, Cin, start, ldo, relu
  gemm_wmma<<<dim3(249, 8),  128, 0, stream>>>(H0, W1T, c1, H1, nullptr, 31872, 512,  256, 1000, 996,  40, 0,  512, 1);
  gemm_wmma<<<dim3(248, 8),  128, 0, stream>>>(H1, W2T, c2, H2, nullptr, 31744, 512, 2560,  996, 992, 512, 0,  512, 1);
  gemm_wmma<<<dim3(247, 8),  128, 0, stream>>>(H2, W3T, c3, H3, nullptr, 31552, 512, 3584,  992, 986, 512, 0,  512, 1);
  gemm_wmma<<<dim3(247, 8),  128, 0, stream>>>(H3, W4T, c4, H4, nullptr, 31520, 512,  512,  986, 985, 512, 1,  512, 1);
  gemm_wmma<<<dim3(246, 24), 128, 0, stream>>>(H4, W5T, c5, H5, nullptr, 31488, 1500, 512,  985, 984, 512, 1, 1500, 1);

  // --- stats pooling ---
  stats_kernel<<<g1(32 * 1504), PB, 0, stream>>>(H5, ST);

  // --- FC head ---
  gemm_wmma<<<dim3(1, 8),  128, 0, stream>>>(ST, F1T, d1, E1, nullptr,  32, 512, 3008, 1, 1, 3008, 0,  512, 1);
  gemm_wmma<<<dim3(1, 8),  128, 0, stream>>>(E1, F2T, d2, EM, emb_out,  32, 512,  512, 1, 1,  512, 0,  512, 1);
  gemm_wmma<<<dim3(1, 16), 128, 0, stream>>>(EM, FLT, dl, nullptr, prob_out, 32, 1000, 512, 1, 1, 512, 0, 1000, 0);
}
